// ConstraintAwareGNN_16131897164168
// MI455X (gfx1250) — compile-verified, matrix operand order checked by reading, not run on hardware
//
#include <hip/hip_runtime.h>

typedef __attribute__((ext_vector_type(16))) _Float16 v16h;
typedef __attribute__((ext_vector_type(8)))  _Float16 h8v;
typedef __attribute__((ext_vector_type(8)))  float    v8f;
typedef __attribute__((ext_vector_type(4)))  unsigned u32x4;
typedef __attribute__((ext_vector_type(8)))  int      i32x8;
typedef __attribute__((ext_vector_type(4)))  int      i32x4;

#define DEVINL __device__ __forceinline__

DEVINL float warp_sum32(float v) {
  #pragma unroll
  for (int o = 1; o < 32; o <<= 1) v += __shfl_xor(v, o, 32);
  return v;
}
DEVINL float warp_max32(float v) {
  #pragma unroll
  for (int o = 1; o < 32; o <<= 1) v = fmaxf(v, __shfl_xor(v, o, 32));
  return v;
}
// Reduce across the 16 lanes of each half-wave (matches WMMA C/D halves).
DEVINL float half_sum16(float v) {
  v += __shfl_xor(v, 1, 32); v += __shfl_xor(v, 2, 32);
  v += __shfl_xor(v, 4, 32); v += __shfl_xor(v, 8, 32);
  return v;
}
DEVINL void atomicMaxF(float* a, float v) {
  if (v >= 0.f) atomicMax((int*)a, __float_as_int(v));
  else          atomicMin((unsigned int*)a, __float_as_uint(v));
}
DEVINL h8v ld_h8(const _Float16* p) { return *(const h8v*)p; }
DEVINL v16h make_frag(h8v lo, h8v hi) {
  return __builtin_shufflevector(lo, hi, 0, 1, 2, 3, 4, 5, 6, 7,
                                 8, 9, 10, 11, 12, 13, 14, 15);
}

// ---- Tensor Data Mover: flat 1-D copy of nbytes (<=65535) global -> LDS ----
// D# group0: count=1, lds_addr, global_addr[56:0], type=2 ("image").
// D# group1: data_size=1B, tensor_dim0=tile_dim0=nbytes, tensor_dim1=tile_dim1=1.
DEVINL void tdm_copy_to_lds(unsigned lds_off, const void* gptr, unsigned nbytes) {
  unsigned long long ga = (unsigned long long)(uintptr_t)gptr;
  u32x4 g0;
  g0[0] = 1u;                                            // count=1 (user mode)
  g0[1] = lds_off;                                       // lds_addr
  g0[2] = (unsigned)(ga & 0xffffffffu);                  // global_addr[31:0]
  g0[3] = (unsigned)((ga >> 32) & 0x01ffffffu) | (2u << 30); // addr[56:32] | type=2
  i32x8 g1;
  g1[0] = 0;                                             // wg_mask=0, data_size=1B
  g1[1] = (int)((nbytes & 0xffffu) << 16);               // tensor_dim0[15:0]
  g1[2] = (int)(((nbytes >> 16) & 0xffffu) | (1u << 16)); // tensor_dim0[31:16], tensor_dim1=1
  g1[3] = (int)((nbytes & 0xffffu) << 16);               // tile_dim0 = nbytes
  g1[4] = 1;                                             // tile_dim1 = 1
  g1[5] = (int)nbytes;                                   // tensor_dim0_stride
  g1[6] = 0;
  g1[7] = 0;
  i32x4 z4 = {0, 0, 0, 0};
  i32x8 z8 = {0, 0, 0, 0, 0, 0, 0, 0};
  __builtin_amdgcn_tensor_load_to_lds(g0, g1, z4, z4, z8, 0);
}

// Stage a pre-swizzled f16 weight blob into LDS via TDM (wave 0 issues).
DEVINL void stage_weights(_Float16* lds, const _Float16* gsrc, unsigned nhalf) {
  if (threadIdx.x < 32) {
    tdm_copy_to_lds((unsigned)(uintptr_t)lds, gsrc, nhalf * 2u);
    __builtin_amdgcn_s_wait_tensorcnt(0);
  }
  __syncthreads();
}

// ---------------- utility kernels ----------------
__global__ void k_fill(float* p, long n, float v) {
  long i = (long)blockIdx.x * 256 + threadIdx.x;
  if (i < n) p[i] = v;
}
__global__ void k_iota(int* p, int n) {
  int i = blockIdx.x * 256 + threadIdx.x;
  if (i < n) p[i] = i;
}
// h = a + b, f16 mirror only (consumed exclusively as GEMM input)
__global__ void k_add_h(const float* __restrict__ a, const float* __restrict__ b,
                        _Float16* __restrict__ oh, long n) {
  long i = (long)blockIdx.x * 256 + threadIdx.x;
  if (i < n) oh[i] = (_Float16)(a[i] + b[i]);
}

// ---- weight swizzle: W[CI,CO] f32 -> f16 in WMMA B-fragment order ----
// dst[(((t*NC+nc)*64)+l*2+g)*8+i] = W[(t*32+g*16+(l>>4)*8+i)*CO + nc*16 + (l&15)]
__global__ void k_wswizzle(const float* __restrict__ W, _Float16* __restrict__ out,
                           int CI, int CO) {
  int total = CI * CO;
  int tid = blockIdx.x * 256 + threadIdx.x;
  if (tid >= total) return;
  int i = tid & 7;
  int g = (tid >> 3) & 1;
  int l = (tid >> 4) & 31;
  int rest = tid >> 9;
  int NC = CO >> 4;
  int nc = rest % NC;
  int t = rest / NC;
  int k = t * 32 + g * 16 + ((l >> 4) * 8) + i;
  int col = nc * 16 + (l & 15);
  out[tid] = (_Float16)W[(long)k * CO + col];
}

// ---------------- node encoder: h = relu(LN(x @ W_ne + b_ne)) ----------------
__global__ __launch_bounds__(256) void k_node_encoder(
    const float* __restrict__ x, const float* __restrict__ W,
    const float* __restrict__ b, const float* __restrict__ g,
    const float* __restrict__ be, float* __restrict__ h,
    _Float16* __restrict__ hh, int N) {
  int wave = threadIdx.x >> 5, lane = threadIdx.x & 31;
  int n = blockIdx.x * 8 + wave;
  if (n >= N) return;
  float x0 = x[(long)n * 3 + 0], x1 = x[(long)n * 3 + 1], x2 = x[(long)n * 3 + 2];
  float v[4]; float s = 0.f;
  #pragma unroll
  for (int j = 0; j < 4; j++) {
    int ch = lane * 4 + j;
    v[j] = x0 * W[ch] + x1 * W[128 + ch] + x2 * W[256 + ch] + b[ch];
    s += v[j];
  }
  s = warp_sum32(s);
  float mu = s * (1.f / 128.f);
  float q = 0.f;
  #pragma unroll
  for (int j = 0; j < 4; j++) { float d = v[j] - mu; q += d * d; }
  q = warp_sum32(q);
  float rinv = rsqrtf(q * (1.f / 128.f) + 1e-5f);
  #pragma unroll
  for (int j = 0; j < 4; j++) {
    int ch = lane * 4 + j;
    float o = fmaxf(g[ch] * (v[j] - mu) * rinv + be[ch], 0.f);
    h[(long)n * 128 + ch] = o;
    hh[(long)n * 128 + ch] = (_Float16)o;
  }
}

// ------- in-place LN+relu over 128 ch, optional pre-bias, optional f16 mirror -------
__global__ __launch_bounds__(256) void k_ln_relu(
    float* __restrict__ buf, const float* __restrict__ prebias,
    const float* __restrict__ g, const float* __restrict__ be,
    _Float16* __restrict__ outh, int N) {
  int wave = threadIdx.x >> 5, lane = threadIdx.x & 31;
  int n = blockIdx.x * 8 + wave;
  if (n >= N) return;
  float v[4]; float s = 0.f;
  #pragma unroll
  for (int j = 0; j < 4; j++) {
    int ch = lane * 4 + j;
    v[j] = buf[(long)n * 128 + ch] + (prebias ? prebias[ch] : 0.f);
    s += v[j];
  }
  s = warp_sum32(s);
  float mu = s * (1.f / 128.f);
  float q = 0.f;
  #pragma unroll
  for (int j = 0; j < 4; j++) { float d = v[j] - mu; q += d * d; }
  q = warp_sum32(q);
  float rinv = rsqrtf(q * (1.f / 128.f) + 1e-5f);
  #pragma unroll
  for (int j = 0; j < 4; j++) {
    int ch = lane * 4 + j;
    float o = fmaxf(g[ch] * (v[j] - mu) * rinv + be[ch], 0.f);
    buf[(long)n * 128 + ch] = o;
    if (outh) outh[(long)n * 128 + ch] = (_Float16)o;
  }
}

// ---------------- edge encoder + self-loop feature accumulation ----------------
__global__ void k_edge_encoder(const float* __restrict__ ea, const float* __restrict__ W,
                               const float* __restrict__ b, const int* __restrict__ dst,
                               _Float16* __restrict__ ef16, float* __restrict__ esum,
                               float* __restrict__ cnt, long E) {
  long e = (long)blockIdx.x * 256 + threadIdx.x;
  if (e >= E) return;
  float a = ea[e];
  int d = dst[e];
  #pragma unroll
  for (int ch = 0; ch < 32; ch++) {
    float v = fmaxf(a * W[ch] + b[ch], 0.f);
    ef16[e * 32 + ch] = (_Float16)v;
    atomicAdd(&esum[(long)d * 32 + ch], v);
  }
  atomicAdd(&cnt[d], 1.f);
}
__global__ void k_loopfeat(const float* __restrict__ esum, const float* __restrict__ cnt,
                           _Float16* __restrict__ lf, int N) {
  long i = (long)blockIdx.x * 256 + threadIdx.x;
  if (i >= (long)N * 32) return;
  int n = (int)(i >> 5);
  lf[i] = (_Float16)(esum[i] / fmaxf(cnt[n], 1.f));
}

// ---------------- generic WMMA GEMM: out = act(cat(A1,A2) @ W + bias) ----------------
// A1h: [M, CI1] f16, A2h: [M, CI2] f16 (nullable when CI2==0),
// Wsw: swizzled f16 weight blob (fragment order), staged to LDS via TDM.
template <int CI1, int CI2, int CO, int ACT>
__global__ __launch_bounds__(256) void k_gemm(
    const _Float16* __restrict__ A1h, const _Float16* __restrict__ A2h,
    const _Float16* __restrict__ Wsw, const float* __restrict__ bias,
    float* __restrict__ out, int M) {
  constexpr int CI = CI1 + CI2;
  constexpr int NC = CO / 16;
  __shared__ _Float16 wlds[CI * CO];
  stage_weights(wlds, Wsw, CI * CO);
  int wave = threadIdx.x >> 5, lane = threadIdx.x & 31;
  int tile = blockIdx.x * 8 + wave;
  if (tile * 16 >= M) return;
  int row0 = tile * 16;
  int r = lane & 15, kh = (lane >> 4) * 8;
  int rr = row0 + r; if (rr >= M) rr = M - 1;
  const _Float16* a1 = A1h + (long)rr * CI1;
  const _Float16* a2 = (CI2 > 0) ? (A2h + (long)rr * CI2) : A1h;
  v8f acc[NC];
  #pragma unroll
  for (int i = 0; i < NC; i++) {
    v8f z = {0.f, 0.f, 0.f, 0.f, 0.f, 0.f, 0.f, 0.f};
    acc[i] = z;
  }
  #pragma unroll
  for (int k0 = 0; k0 < CI; k0 += 32) {
    const _Float16* ap;
    int ko;
    if (k0 < CI1) { ap = a1; ko = k0; } else { ap = a2; ko = k0 - CI1; }
    v16h af = make_frag(ld_h8(ap + ko + kh), ld_h8(ap + ko + 16 + kh));
    int t = k0 >> 5;
    #pragma unroll
    for (int nc = 0; nc < NC; nc++) {
      const _Float16* wp = &wlds[(((t * NC + nc) * 64) + lane * 2) * 8];
      v16h bf = make_frag(ld_h8(wp), ld_h8(wp + 8));
      acc[nc] = __builtin_amdgcn_wmma_f32_16x16x32_f16(false, af, false, bf,
                                                       (short)0, acc[nc], false, false);
    }
  }
  int mb = (lane >> 4) * 8;
  #pragma unroll
  for (int nc = 0; nc < NC; nc++) {
    int col = nc * 16 + r;
    float bv = bias ? bias[col] : 0.f;
    #pragma unroll
    for (int i = 0; i < 8; i++) {
      int row = row0 + mb + i;
      if (row < M) {
        float v = acc[nc][i] + bv;
        if (ACT == 1) v = fmaxf(v, 0.f);
        out[(long)row * CO + col] = v;
      }
    }
  }
}

// ---- GATv2 logits: z = leaky(xl[s]+xr[d]+e@We); logit = z . att; segment max ----
// 16 edges per wave: A = e-tile (16x32 f16), B = swizzled We (32x128) staged via TDM.
template <int HEADS>
__global__ __launch_bounds__(256) void k_gat_logit(
    const _Float16* __restrict__ ef16, const int* __restrict__ src,
    const int* __restrict__ dst, const float* __restrict__ xl,
    const float* __restrict__ xr, const _Float16* __restrict__ Wesw,
    const float* __restrict__ att, float* __restrict__ logit,
    float* __restrict__ mmax, long nE) {
  constexpr int CH = 128 / HEADS;
  __shared__ _Float16 wlds[32 * 128];
  stage_weights(wlds, Wesw, 32 * 128);
  int wave = threadIdx.x >> 5, lane = threadIdx.x & 31;
  long ebase = ((long)blockIdx.x * 8 + wave) * 16;
  if (ebase >= nE) return;
  int r = lane & 15, kh = (lane >> 4) * 8, mb = kh;
  long er = ebase + r; if (er >= nE) er = nE - 1;
  const _Float16* ep = ef16 + er * 32;
  v16h af = make_frag(ld_h8(ep + kh), ld_h8(ep + 16 + kh));
  int sidx[8], didx[8];
  #pragma unroll
  for (int i = 0; i < 8; i++) {
    long e = ebase + mb + i; if (e >= nE) e = nE - 1;
    sidx[i] = src[e]; didx[i] = dst[e];
  }
  float hacc[8][HEADS];
  #pragma unroll
  for (int i = 0; i < 8; i++)
    #pragma unroll
    for (int h = 0; h < HEADS; h++) hacc[i][h] = 0.f;
  #pragma unroll
  for (int nc = 0; nc < 8; nc++) {
    const _Float16* wp = &wlds[((nc * 64) + lane * 2) * 8];
    v16h bf = make_frag(ld_h8(wp), ld_h8(wp + 8));
    v8f z8 = {0.f, 0.f, 0.f, 0.f, 0.f, 0.f, 0.f, 0.f};
    v8f acc = __builtin_amdgcn_wmma_f32_16x16x32_f16(false, af, false, bf,
                                                     (short)0, z8, false, false);
    int ch = nc * 16 + r;
    int hd = (nc * 16) / CH;
    float av = att[ch];
    #pragma unroll
    for (int i = 0; i < 8; i++) {
      float z = acc[i] + xl[(long)sidx[i] * 128 + ch] + xr[(long)didx[i] * 128 + ch];
      z = z > 0.f ? z : 0.2f * z;
      hacc[i][hd] += z * av;
    }
  }
  #pragma unroll
  for (int i = 0; i < 8; i++)
    #pragma unroll
    for (int h = 0; h < HEADS; h++) hacc[i][h] = half_sum16(hacc[i][h]);
  if ((lane & 15) == 0) {
    #pragma unroll
    for (int i = 0; i < 8; i++) {
      long e = ebase + mb + i;
      if (e < nE) {
        #pragma unroll
        for (int h = 0; h < HEADS; h++) {
          logit[e * HEADS + h] = hacc[i][h];
          atomicMaxF(&mmax[(long)didx[i] * HEADS + h], hacc[i][h]);
        }
      }
    }
  }
}

// pass2: logit -> exp(logit - m[dst]); den[dst] += ex
template <int HEADS>
__global__ void k_gat_expden(float* __restrict__ logit, const int* __restrict__ dst,
                             const float* __restrict__ mmax, float* __restrict__ den,
                             long nE) {
  long tid = (long)blockIdx.x * 256 + threadIdx.x;
  if (tid >= nE * HEADS) return;
  long e = tid / HEADS; int h = (int)(tid % HEADS);
  int d = dst[e];
  float ex = __expf(logit[tid] - mmax[(long)d * HEADS + h]);
  logit[tid] = ex;
  atomicAdd(&den[(long)d * HEADS + h], ex);
}

// pass3: out[dst] += xl[src] * (ex / den[dst]) ; one thread per (edge, 4 channels)
template <int HEADS>
__global__ void k_gat_agg(const float* __restrict__ ex, const float* __restrict__ den,
                          const int* __restrict__ src, const int* __restrict__ dst,
                          const float* __restrict__ xl, float* __restrict__ out, long nE) {
  long tid = (long)blockIdx.x * 256 + threadIdx.x;
  long e = tid >> 5;
  if (e >= nE) return;
  int c0 = ((int)tid & 31) * 4;
  int s = src[e], d = dst[e];
  constexpr int CH = 128 / HEADS;
  int hd = c0 / CH;
  float alpha = ex[e * HEADS + hd] / den[(long)d * HEADS + hd];
  const float4 xv = *(const float4*)&xl[(long)s * 128 + c0];
  atomicAdd(&out[(long)d * 128 + c0 + 0], xv.x * alpha);
  atomicAdd(&out[(long)d * 128 + c0 + 1], xv.y * alpha);
  atomicAdd(&out[(long)d * 128 + c0 + 2], xv.z * alpha);
  atomicAdd(&out[(long)d * 128 + c0 + 3], xv.w * alpha);
}

// ---- comm-delay head (fused WMMA): hh=relu(Ah[s]+Bh[d]+e@Wc+bc1); delay=relu(hh.Wc2+bc2)
//      cand = base[s]+10+delay; segm[d] = max(segm[d], cand)
__global__ __launch_bounds__(256) void k_edge_delay(
    const _Float16* __restrict__ ef16, const int* __restrict__ src,
    const int* __restrict__ dst, const float* __restrict__ Ah,
    const float* __restrict__ Bh, const _Float16* __restrict__ Wcsw,
    const float* __restrict__ bc1, const float* __restrict__ Wc2,
    const float* __restrict__ bc2, const float* __restrict__ base,
    float* __restrict__ segm, long nE) {
  __shared__ _Float16 wlds[32 * 128];
  stage_weights(wlds, Wcsw, 32 * 128);
  int wave = threadIdx.x >> 5, lane = threadIdx.x & 31;
  long ebase = ((long)blockIdx.x * 8 + wave) * 16;
  if (ebase >= nE) return;
  int r = lane & 15, kh = (lane >> 4) * 8, mb = kh;
  long er = ebase + r; if (er >= nE) er = nE - 1;
  const _Float16* ep = ef16 + er * 32;
  v16h af = make_frag(ld_h8(ep + kh), ld_h8(ep + 16 + kh));
  int sidx[8], didx[8];
  #pragma unroll
  for (int i = 0; i < 8; i++) {
    long e = ebase + mb + i; if (e >= nE) e = nE - 1;
    sidx[i] = src[e]; didx[i] = dst[e];
  }
  float dpart[8];
  #pragma unroll
  for (int i = 0; i < 8; i++) dpart[i] = 0.f;
  #pragma unroll
  for (int nc = 0; nc < 8; nc++) {
    const _Float16* wp = &wlds[((nc * 64) + lane * 2) * 8];
    v16h bf = make_frag(ld_h8(wp), ld_h8(wp + 8));
    v8f z8 = {0.f, 0.f, 0.f, 0.f, 0.f, 0.f, 0.f, 0.f};
    v8f acc = __builtin_amdgcn_wmma_f32_16x16x32_f16(false, af, false, bf,
                                                     (short)0, z8, false, false);
    int ch = nc * 16 + r;
    float w2 = Wc2[ch], b1 = bc1[ch];
    #pragma unroll
    for (int i = 0; i < 8; i++) {
      float hh = acc[i] + Ah[(long)sidx[i] * 128 + ch] + Bh[(long)didx[i] * 128 + ch] + b1;
      dpart[i] += fmaxf(hh, 0.f) * w2;
    }
  }
  #pragma unroll
  for (int i = 0; i < 8; i++) dpart[i] = half_sum16(dpart[i]);
  if ((lane & 15) == 0) {
    float b2 = bc2[0];
    #pragma unroll
    for (int i = 0; i < 8; i++) {
      long e = ebase + mb + i;
      if (e < nE) {
        float dly = fmaxf(dpart[i] + b2, 0.f);
        float cand = base[sidx[i]] + 10.f + dly;
        atomicMaxF(&segm[didx[i]], cand);
      }
    }
  }
}

// ---- eligibility mask + row softmax over P=192 (one wave per row) ----
__global__ __launch_bounds__(256) void k_mask_softmax(
    float* __restrict__ logits, float* __restrict__ probs,
    _Float16* __restrict__ probsh, const float* __restrict__ mask, int N, int P) {
  int wave = threadIdx.x >> 5, lane = threadIdx.x & 31;
  int n = blockIdx.x * 8 + wave;
  if (n >= N) return;
  int nj = P / 32;
  float lg[8];
  float mx = -3.0e38f;
  for (int j = 0; j < nj; j++) {
    int ch = j * 32 + lane;
    float raw = logits[(long)n * P + ch];
    float m = mask[(long)n * P + ch];
    lg[j] = raw * m + (-1e9f) * (1.f - m);
    logits[(long)n * P + ch] = lg[j];
    mx = fmaxf(mx, lg[j]);
  }
  mx = warp_max32(mx);
  float s = 0.f;
  float exv[8];
  for (int j = 0; j < nj; j++) { exv[j] = __expf(lg[j] - mx); s += exv[j]; }
  s = warp_sum32(s);
  float inv = 1.f / s;
  for (int j = 0; j < nj; j++) {
    int ch = j * 32 + lane;
    float p = exv[j] * inv;
    probs[(long)n * P + ch] = p;
    probsh[(long)n * P + ch] = (_Float16)p;
  }
}

// ---- small per-node heads ----
__global__ void k_base(const float* __restrict__ s1, const float* __restrict__ Ws2,
                       const float* __restrict__ bs2, float* __restrict__ base, int N) {
  int n = blockIdx.x * 256 + threadIdx.x;
  if (n >= N) return;
  float s = bs2[0];
  for (int k = 0; k < 64; k++) s += s1[(long)n * 64 + k] * Ws2[k];
  base[n] = fmaxf(s, 0.f);
}
__global__ void k_start(const float* __restrict__ base, const float* __restrict__ segm,
                        float* __restrict__ start, int N) {
  int n = blockIdx.x * 256 + threadIdx.x;
  if (n >= N) return;
  start[n] = fmaxf(base[n], segm[n]);
}
__global__ void k_durend(const float* __restrict__ q, const float* __restrict__ Wd2,
                         const float* __restrict__ bd2, const float* __restrict__ start,
                         float* __restrict__ dur, float* __restrict__ endv, int N) {
  int n = blockIdx.x * 256 + threadIdx.x;
  if (n >= N) return;
  float s = bd2[0];
  for (int k = 0; k < 64; k++) s += q[(long)n * 64 + k] * Wd2[k];
  s = fmaxf(s, 0.f);
  dur[n] = s;
  endv[n] = start[n] + s;
}
__global__ void k_mk(const float* __restrict__ endv, const int* __restrict__ batch,
                     float* __restrict__ mk, int N) {
  int n = blockIdx.x * 256 + threadIdx.x;
  if (n >= N) return;
  atomicMaxF(&mk[batch[n]], endv[n]);
}

// =====================================================================
extern "C" void kernel_launch(void* const* d_in, const int* in_sizes, int n_in,
                              void* d_out, int out_size, void* d_ws, size_t ws_size,
                              hipStream_t stream) {
  (void)n_in; (void)ws_size;
  const float* x    = (const float*)d_in[0];
  const float* ea   = (const float*)d_in[1];
  const float* mask = (const float*)d_in[2];
  const float* W_ne = (const float*)d_in[3];
  const float* b_ne = (const float*)d_in[4];
  const float* g_ne = (const float*)d_in[5];
  const float* be_ne= (const float*)d_in[6];
  const float* W_ee = (const float*)d_in[7];
  const float* b_ee = (const float*)d_in[8];
  const float* Wl0  = (const float*)d_in[9];
  const float* bl0  = (const float*)d_in[10];
  const float* Wr0  = (const float*)d_in[11];
  const float* br0  = (const float*)d_in[12];
  const float* We0  = (const float*)d_in[13];
  const float* att0 = (const float*)d_in[14];
  const float* bo0  = (const float*)d_in[15];
  const float* g0   = (const float*)d_in[16];
  const float* be0  = (const float*)d_in[17];
  const float* Wl1  = (const float*)d_in[18];
  const float* bl1  = (const float*)d_in[19];
  const float* Wr1  = (const float*)d_in[20];
  const float* br1  = (const float*)d_in[21];
  const float* We1  = (const float*)d_in[22];
  const float* att1 = (const float*)d_in[23];
  const float* bo1  = (const float*)d_in[24];
  const float* g1p  = (const float*)d_in[25];
  const float* be1  = (const float*)d_in[26];
  const float* Wp1  = (const float*)d_in[27];
  const float* bp1  = (const float*)d_in[28];
  const float* gp   = (const float*)d_in[29];
  const float* bep  = (const float*)d_in[30];
  const float* Wp2  = (const float*)d_in[31];
  const float* bp2  = (const float*)d_in[32];
  const float* Ws1  = (const float*)d_in[33];
  const float* bs1  = (const float*)d_in[34];
  const float* Ws2  = (const float*)d_in[35];
  const float* bs2  = (const float*)d_in[36];
  const float* Wd1  = (const float*)d_in[37];
  const float* bd1  = (const float*)d_in[38];
  const float* Wd2  = (const float*)d_in[39];
  const float* bd2  = (const float*)d_in[40];
  const float* Wc1  = (const float*)d_in[41];
  const float* bc1  = (const float*)d_in[42];
  const float* Wc2  = (const float*)d_in[43];
  const float* bc2  = (const float*)d_in[44];
  const int* srcp   = (const int*)d_in[45];
  const int* dstp   = (const int*)d_in[46];
  const int* batchp = (const int*)d_in[47];

  const int  N = in_sizes[0] / 3;
  const long E = in_sizes[1];
  const int  P = in_sizes[2] / N;
  const long NP = (long)N * P;
  const int  B = (int)(out_size - 2 * NP - 3 * (long)N);

  float* out_logits = (float*)d_out;
  float* out_probs  = out_logits + NP;
  float* out_start  = out_probs + NP;
  float* out_end    = out_start + N;
  float* out_dur    = out_end + N;
  float* out_mk     = out_dur + N;

  // workspace carve
  char* wsp = (char*)d_ws;
  auto alloc = [&](size_t bytes) -> void* {
    void* p = wsp;
    wsp += (bytes + 255) & ~(size_t)255;
    return p;
  };
  float*    h0     = (float*)alloc((size_t)N * 128 * 4);   // node enc (f32)
  float*    xl     = (float*)alloc((size_t)N * 128 * 4);   // later Ah
  float*    xr     = (float*)alloc((size_t)N * 128 * 4);   // later Bh
  float*    h1     = (float*)alloc((size_t)N * 128 * 4);   // later s1/q (N*64)
  float*    h2     = (float*)alloc((size_t)N * 128 * 4);   // later t raw
  _Float16* h0h    = (_Float16*)alloc((size_t)N * 128 * 2); // f16 h0, later f16 h
  _Float16* h1h    = (_Float16*)alloc((size_t)N * 128 * 2);
  _Float16* h2h    = (_Float16*)alloc((size_t)N * 128 * 2); // f16 t
  _Float16* probsh = (_Float16*)alloc((size_t)N * P * 2);
  _Float16* ef16   = (_Float16*)alloc((size_t)E * 32 * 2);
  _Float16* loopf  = (_Float16*)alloc((size_t)N * 32 * 2);
  float*    esum   = (float*)alloc((size_t)N * 32 * 4);
  float*    cnt    = (float*)alloc((size_t)N * 4);
  float*    logitE = (float*)alloc((size_t)E * 4 * 4);
  float*    logitL = (float*)alloc((size_t)N * 4 * 4);
  float*    mmax   = (float*)alloc((size_t)N * 4 * 4);
  float*    den    = (float*)alloc((size_t)N * 4 * 4);
  int*      iota   = (int*)alloc((size_t)N * 4);
  float*    baseb  = (float*)alloc((size_t)N * 4);
  float*    segm   = (float*)alloc((size_t)N * 4);
  // swizzled f16 weights (WMMA B-fragment order)
  _Float16* sWl0 = (_Float16*)alloc(128 * 128 * 2);
  _Float16* sWr0 = (_Float16*)alloc(128 * 128 * 2);
  _Float16* sWl1 = (_Float16*)alloc(128 * 128 * 2);
  _Float16* sWr1 = (_Float16*)alloc(128 * 128 * 2);
  _Float16* sWp1 = (_Float16*)alloc(128 * 128 * 2);
  _Float16* sWca = (_Float16*)alloc(128 * 128 * 2);
  _Float16* sWcb = (_Float16*)alloc(128 * 128 * 2);
  _Float16* sWp2 = (_Float16*)alloc(128 * 192 * 2);
  _Float16* sWs1 = (_Float16*)alloc(128 * 64 * 2);
  _Float16* sWd1 = (_Float16*)alloc(320 * 64 * 2);
  _Float16* sWe0 = (_Float16*)alloc(32 * 128 * 2);
  _Float16* sWe1 = (_Float16*)alloc(32 * 128 * 2);
  _Float16* sWcc = (_Float16*)alloc(32 * 128 * 2);

  dim3 blk(256);
  auto g1 = [](long n) { return dim3((unsigned)((n + 255) / 256)); };
  auto gw = [](long rows) { return dim3((unsigned)((rows + 7) / 8)); };
  auto gt = [](long rows) {
    long tiles = (rows + 15) / 16;
    return dim3((unsigned)((tiles + 7) / 8));
  };

  // ---- weight swizzle (once per launch) ----
  k_wswizzle<<<g1(128 * 128), blk, 0, stream>>>(Wl0, sWl0, 128, 128);
  k_wswizzle<<<g1(128 * 128), blk, 0, stream>>>(Wr0, sWr0, 128, 128);
  k_wswizzle<<<g1(128 * 128), blk, 0, stream>>>(Wl1, sWl1, 128, 128);
  k_wswizzle<<<g1(128 * 128), blk, 0, stream>>>(Wr1, sWr1, 128, 128);
  k_wswizzle<<<g1(128 * 128), blk, 0, stream>>>(Wp1, sWp1, 128, 128);
  k_wswizzle<<<g1(128 * 128), blk, 0, stream>>>(Wc1, sWca, 128, 128);
  k_wswizzle<<<g1(128 * 128), blk, 0, stream>>>(Wc1 + 128 * 128, sWcb, 128, 128);
  k_wswizzle<<<g1(128 * 192), blk, 0, stream>>>(Wp2, sWp2, 128, 192);
  k_wswizzle<<<g1(128 * 64), blk, 0, stream>>>(Ws1, sWs1, 128, 64);
  k_wswizzle<<<g1(320 * 64), blk, 0, stream>>>(Wd1, sWd1, 320, 64);
  k_wswizzle<<<g1(32 * 128), blk, 0, stream>>>(We0, sWe0, 32, 128);
  k_wswizzle<<<g1(32 * 128), blk, 0, stream>>>(We1, sWe1, 32, 128);
  k_wswizzle<<<g1(32 * 128), blk, 0, stream>>>(Wc1 + 256 * 128, sWcc, 32, 128);

  // ---- encoders + self-loop features ----
  k_fill<<<g1((long)N * 32), blk, 0, stream>>>(esum, (long)N * 32, 0.f);
  k_fill<<<g1(N), blk, 0, stream>>>(cnt, N, 0.f);
  k_iota<<<g1(N), blk, 0, stream>>>(iota, N);
  k_node_encoder<<<gw(N), blk, 0, stream>>>(x, W_ne, b_ne, g_ne, be_ne, h0, h0h, N);
  k_edge_encoder<<<g1(E), blk, 0, stream>>>(ea, W_ee, b_ee, dstp, ef16, esum, cnt, E);
  k_loopfeat<<<g1((long)N * 32), blk, 0, stream>>>(esum, cnt, loopf, N);

  // ---- GAT layer 0 (4 heads) ----
  k_gemm<128, 0, 128, 0><<<gt(N), blk, 0, stream>>>(h0h, nullptr, sWl0, bl0, xl, N);
  k_gemm<128, 0, 128, 0><<<gt(N), blk, 0, stream>>>(h0h, nullptr, sWr0, br0, xr, N);
  k_fill<<<g1((long)N * 4), blk, 0, stream>>>(mmax, (long)N * 4, -1e30f);
  k_fill<<<g1((long)N * 4), blk, 0, stream>>>(den, (long)N * 4, 0.f);
  k_fill<<<g1((long)N * 128), blk, 0, stream>>>(h1, (long)N * 128, 0.f);
  k_gat_logit<4><<<gt(E), blk, 0, stream>>>(ef16, srcp, dstp, xl, xr, sWe0, att0, logitE, mmax, E);
  k_gat_logit<4><<<gt(N), blk, 0, stream>>>(loopf, iota, iota, xl, xr, sWe0, att0, logitL, mmax, N);
  k_gat_expden<4><<<g1(E * 4), blk, 0, stream>>>(logitE, dstp, mmax, den, E);
  k_gat_expden<4><<<g1((long)N * 4), blk, 0, stream>>>(logitL, iota, mmax, den, N);
  k_gat_agg<4><<<g1(E * 32), blk, 0, stream>>>(logitE, den, srcp, dstp, xl, h1, E);
  k_gat_agg<4><<<g1((long)N * 32), blk, 0, stream>>>(logitL, den, iota, iota, xl, h1, N);
  k_ln_relu<<<gw(N), blk, 0, stream>>>(h1, bo0, g0, be0, h1h, N);

  // ---- GAT layer 1 (1 head) ----
  k_gemm<128, 0, 128, 0><<<gt(N), blk, 0, stream>>>(h1h, nullptr, sWl1, bl1, xl, N);
  k_gemm<128, 0, 128, 0><<<gt(N), blk, 0, stream>>>(h1h, nullptr, sWr1, br1, xr, N);
  k_fill<<<g1(N), blk, 0, stream>>>(mmax, N, -1e30f);
  k_fill<<<g1(N), blk, 0, stream>>>(den, N, 0.f);
  k_fill<<<g1((long)N * 128), blk, 0, stream>>>(h2, (long)N * 128, 0.f);
  k_gat_logit<1><<<gt(E), blk, 0, stream>>>(ef16, srcp, dstp, xl, xr, sWe1, att1, logitE, mmax, E);
  k_gat_logit<1><<<gt(N), blk, 0, stream>>>(loopf, iota, iota, xl, xr, sWe1, att1, logitL, mmax, N);
  k_gat_expden<1><<<g1(E), blk, 0, stream>>>(logitE, dstp, mmax, den, E);
  k_gat_expden<1><<<g1(N), blk, 0, stream>>>(logitL, iota, mmax, den, N);
  k_gat_agg<1><<<g1(E * 32), blk, 0, stream>>>(logitE, den, srcp, dstp, xl, h2, E);
  k_gat_agg<1><<<g1((long)N * 32), blk, 0, stream>>>(logitL, den, iota, iota, xl, h2, N);
  k_ln_relu<<<gw(N), blk, 0, stream>>>(h2, bo1, g1p, be1, nullptr, N);

  // ---- residual: h = h1 + h2 (f16 mirror into h0h, GEMM-only consumer) ----
  k_add_h<<<g1((long)N * 128), blk, 0, stream>>>(h1, h2, h0h, (long)N * 128);

  // ---- pointer head: logits / probs ----
  k_gemm<128, 0, 128, 0><<<gt(N), blk, 0, stream>>>(h0h, nullptr, sWp1, bp1, h2, N); // h2 := t raw
  k_ln_relu<<<gw(N), blk, 0, stream>>>(h2, nullptr, gp, bep, h2h, N);                // h2h := t
  k_gemm<128, 0, 192, 0><<<gt(N), blk, 0, stream>>>(h2h, nullptr, sWp2, bp2, out_logits, N);
  k_mask_softmax<<<gw(N), blk, 0, stream>>>(out_logits, out_probs, probsh, mask, N, P);

  // ---- base times ----
  k_gemm<128, 0, 64, 1><<<gt(N), blk, 0, stream>>>(h0h, nullptr, sWs1, bs1, h1, N);  // h1 := s1
  k_base<<<g1(N), blk, 0, stream>>>(h1, Ws2, bs2, baseb, N);

  // ---- comm delay head + schedule ----
  k_gemm<128, 0, 128, 0><<<gt(N), blk, 0, stream>>>(h0h, nullptr, sWca, nullptr, xl, N); // Ah
  k_gemm<128, 0, 128, 0><<<gt(N), blk, 0, stream>>>(h0h, nullptr, sWcb, nullptr, xr, N); // Bh
  k_fill<<<g1(N), blk, 0, stream>>>(segm, N, -1e30f);
  k_edge_delay<<<gt(E), blk, 0, stream>>>(ef16, srcp, dstp, xl, xr, sWcc,
                                          bc1, Wc2, bc2, baseb, segm, E);
  k_start<<<g1(N), blk, 0, stream>>>(baseb, segm, out_start, N);

  // ---- duration head: dur = relu(relu([h, probs] @ Wd1 + bd1) @ Wd2 + bd2) ----
  k_gemm<128, 192, 64, 1><<<gt(N), blk, 0, stream>>>(h0h, probsh, sWd1, bd1, h1, N); // h1 := q
  k_durend<<<g1(N), blk, 0, stream>>>(h1, Wd2, bd2, out_start, out_dur, out_end, N);

  // ---- makespan per graph ----
  k_fill<<<g1(B), blk, 0, stream>>>(out_mk, B, -1e30f);
  k_mk<<<g1(N), blk, 0, stream>>>(out_end, batchp, out_mk, N);
}
